// MoTAttention_58394375356835
// MI455X (gfx1250) — matrix operand order repaired and checked by reading, not compiled, not run on hardware
//
#include <hip/hip_runtime.h>
#include <hip/hip_bf16.h>

// ---------------------------------------------------------------------------
// MoT attention for MI455X (gfx1250): bf16 WMMA GEMMs + flash attention,
// with Tensor Data Mover (TDM) staging of bf16 tiles into LDS.
// B=2 S=2048 D=2048 H=16 KVH=4 HD=128 N_REP=4
// ---------------------------------------------------------------------------

typedef __attribute__((ext_vector_type(16))) __bf16          v16bf;
typedef __attribute__((ext_vector_type(8)))  float           v8f;
typedef __attribute__((ext_vector_type(8)))  unsigned short  us8;
typedef __attribute__((ext_vector_type(4)))  float           f4;
typedef __attribute__((ext_vector_type(4)))  unsigned int    u4;
typedef __attribute__((ext_vector_type(4)))  unsigned int    v4u;
typedef __attribute__((ext_vector_type(8)))  int             v8i;
typedef __attribute__((ext_vector_type(4)))  int             v4i;

union Frag { v16bf v; us8 h[2]; unsigned short s[16]; };

__device__ __forceinline__ unsigned short f2bf(float f) {   // RNE (cold paths)
    union { float f; unsigned int u; } x; x.f = f;
    unsigned int r = x.u + 0x7FFFu + ((x.u >> 16) & 1u);
    return (unsigned short)(r >> 16);
}
__device__ __forceinline__ unsigned short f2bft(float f) {  // truncate (hot)
    union { float f; unsigned int u; } x; x.f = f;
    return (unsigned short)(x.u >> 16);
}
__device__ __forceinline__ unsigned int pk2bf(float a, float b) { // v_perm_b32
    union { float f; unsigned int u; } xa, xb; xa.f = a; xb.f = b;
    return (xa.u >> 16) | (xb.u & 0xFFFF0000u);
}
__device__ __forceinline__ float bf2f(unsigned short h) {
    union { unsigned int u; float f; } x; x.u = ((unsigned int)h) << 16;
    return x.f;
}
__device__ __forceinline__ unsigned lds_off(const void* p) {
    // generic LDS pointer: low 32 bits are the LDS byte offset (ISA 10.2)
    return (unsigned)(unsigned long long)p;
}

// ---------------------------------------------------------------------------
// TDM: DMA a 2D bf16 tile (dim0 contiguous elements x dim1 rows, row stride
// in elements) from global memory into LDS. D# built per ISA 8.3/8.4.
// Issue from ONE wave; complete with s_wait_tensorcnt 0 before the barrier.
// ---------------------------------------------------------------------------
__device__ __forceinline__ void tdm_load_2d_bf16(unsigned lds_byte_off,
                                                 const unsigned short* gptr,
                                                 unsigned dim0, unsigned dim1,
                                                 unsigned stride_elems)
{
    unsigned long long ga = (unsigned long long)(const void*)gptr;
    v4u g0;
    g0[0] = 1u;                                    // count=1 (valid user D#)
    g0[1] = lds_byte_off;                          // lds_addr
    g0[2] = (unsigned)ga;                          // global_addr[31:0]
    g0[3] = (unsigned)((ga >> 32) & 0x01FFFFFFu)   // global_addr[56:32]
            | (2u << 30);                          // type=2 ("image")
    v8i g1;
    g1[0] = (int)(1u << 16);                       // data_size=1 -> 2 bytes
    g1[1] = (int)(dim0 << 16);                     // tensor_dim0[15:0]
    g1[2] = (int)((dim0 >> 16) | (dim1 << 16));    // tensor_dim0[31:16]|dim1[15:0]
    g1[3] = (int)((dim1 >> 16) | (dim0 << 16));    // tensor_dim1[31:16]|tile_dim0
    g1[4] = (int)(dim1 & 0xFFFFu);                 // tile_dim1 (tile_dim2=0)
    g1[5] = (int)stride_elems;                     // tensor_dim0_stride[31:0]
    g1[6] = 0;                                     // stride hi / dim1_stride
    g1[7] = 0;
    v4i gz = {0, 0, 0, 0};
#if __clang_major__ >= 23
    v8i gz8 = {0, 0, 0, 0, 0, 0, 0, 0};
    __builtin_amdgcn_tensor_load_to_lds(g0, g1, gz, gz, gz8, 0);
#else
    __builtin_amdgcn_tensor_load_to_lds(g0, g1, gz, gz, 0);
#endif
}

// ---------------------------------------------------------------------------
// Dual-expert GEMM: C[r,:] = (mask[r] ? A@W0 : A@W1)[r,:]
// Block tile 128(M) x 32(N), K-step 64 (8 WMMAs per stage), 8 waves.
// LDS: A row-major [m][k]; W tiles transposed [n][k] -> every fragment is
// two contiguous 16B chunks (ds_load_b128 x2). bf16 A tiles staged by TDM.
// ---------------------------------------------------------------------------
template<bool A_BF16, bool OUT_BF16>
__global__ __launch_bounds__(256)
void dual_gemm(const void* __restrict__ Ap,
               const float* __restrict__ W0, const float* __restrict__ W1,
               const unsigned char* __restrict__ mask,
               void* __restrict__ Cp, int M, int N, int K)
{
    __shared__ alignas(16) unsigned short As[128 * 64];      // [m][k]
    __shared__ alignas(16) unsigned short Bs[2][32 * 64];    // [expert][n][k]
    __shared__ unsigned char Ms[128];

    const int tid  = threadIdx.x;
    const int wave = tid >> 5;
    const int lane = tid & 31;
    const int hh   = lane >> 4;
    const int ml   = lane & 15;
    const int row0 = blockIdx.y * 128;
    const int col0 = blockIdx.x * 32;

    if (tid < 128) Ms[tid] = mask[row0 + tid];

    // A staging (f32 path): thread owns 32 contiguous k of one row
    const int ar = tid >> 1;
    const int ac = (tid & 1) * 32;

    v8f acc[2][2] = {};  // [expert][ntile]

    for (int k0 = 0; k0 < K; k0 += 64) {
        __syncthreads();
        // ---- stage A tile (128x64 bf16) ----
        if (A_BF16) {
            if (wave == 0)
                tdm_load_2d_bf16(lds_off(&As[0]),
                                 (const unsigned short*)Ap + (size_t)row0 * K + k0,
                                 64, 128, (unsigned)K);
        } else {
            const float* ap = (const float*)Ap + (size_t)(row0 + ar) * K + k0 + ac;
            #pragma unroll
            for (int q = 0; q < 2; ++q) {
                f4 w0 = ((const f4*)ap)[4 * q + 0];
                f4 w1 = ((const f4*)ap)[4 * q + 1];
                f4 w2 = ((const f4*)ap)[4 * q + 2];
                f4 w3 = ((const f4*)ap)[4 * q + 3];
                u4 plo, phi;
                plo[0] = pk2bf(w0[0], w0[1]); plo[1] = pk2bf(w0[2], w0[3]);
                plo[2] = pk2bf(w1[0], w1[1]); plo[3] = pk2bf(w1[2], w1[3]);
                phi[0] = pk2bf(w2[0], w2[1]); phi[1] = pk2bf(w2[2], w2[3]);
                phi[2] = pk2bf(w3[0], w3[1]); phi[3] = pk2bf(w3[2], w3[3]);
                *(u4*)&As[ar * 64 + ac + q * 16]     = plo;
                *(u4*)&As[ar * 64 + ac + q * 16 + 8] = phi;
            }
        }
        // ---- stage both expert weight tiles transposed: Bs[n][k] ----
        #pragma unroll
        for (int j = 0; j < 8; ++j) {
            int li = tid + j * 256;
            int r = li >> 5, c = li & 31;  // r = k (0..63), c = n (coalesced)
            Bs[0][c * 64 + r] = f2bft(W0[(size_t)(k0 + r) * N + col0 + c]);
            Bs[1][c * 64 + r] = f2bft(W1[(size_t)(k0 + r) * N + col0 + c]);
        }
        if (k0 + 64 < K) {
            __builtin_prefetch(&W0[(size_t)(k0 + 64) * N + col0 + lane], 0, 1);
            __builtin_prefetch(&W1[(size_t)(k0 + 64) * N + col0 + lane], 0, 1);
        }
        if (A_BF16 && wave == 0)
            __builtin_amdgcn_s_wait_tensorcnt((short)0);
        __syncthreads();

        // ---- 8 WMMAs: 2 K-substeps x 2 experts x 2 N-tiles ----
        #pragma unroll
        for (int kb2 = 0; kb2 < 2; ++kb2) {
            Frag a;  // ISA A 16x32: e<8 -> K=8h+e ; e>=8 -> K=16+8h+(e-8)
            const int abase = (wave * 16 + ml) * 64 + kb2 * 32;
            a.h[0] = *(const us8*)&As[abase + 8 * hh];
            a.h[1] = *(const us8*)&As[abase + 16 + 8 * hh];
            #pragma unroll
            for (int ex = 0; ex < 2; ++ex) {
                #pragma unroll
                for (int nt = 0; nt < 2; ++nt) {
                    Frag b;  // B 32x16: lane n, e<8 -> K=8h+e ; e>=8 -> K=16+8h+(e-8)
                    const int bbase = (nt * 16 + ml) * 64 + kb2 * 32;
                    b.h[0] = *(const us8*)&Bs[ex][bbase + 8 * hh];
                    b.h[1] = *(const us8*)&Bs[ex][bbase + 16 + 8 * hh];
                    acc[ex][nt] = __builtin_amdgcn_wmma_f32_16x16x32_bf16(
                        false, a.v, false, b.v, (short)0, acc[ex][nt], false, false);
                }
            }
        }
    }

    // epilogue: per-row expert select, store
    #pragma unroll
    for (int nt = 0; nt < 2; ++nt) {
        int col = col0 + nt * 16 + ml;
        #pragma unroll
        for (int v = 0; v < 8; ++v) {
            int rloc = wave * 16 + v + 8 * hh;
            int row  = row0 + rloc;
            float val = Ms[rloc] ? acc[0][nt][v] : acc[1][nt][v];
            if (OUT_BF16)
                ((unsigned short*)Cp)[(size_t)row * N + col] = f2bf(val);
            else
                ((float*)Cp)[(size_t)row * N + col] = val;
        }
    }
}

// ---------------------------------------------------------------------------
// RoPE over bf16 q (4096x2048) and k (4096x512), interleaved pairs.
// ---------------------------------------------------------------------------
__global__ __launch_bounds__(256)
void rope_kernel(unsigned short* __restrict__ q, unsigned short* __restrict__ k,
                 const float* __restrict__ fc, const float* __restrict__ fs)
{
    const long long QP = (long long)4096 * 1024;
    const long long KP = (long long)4096 * 256;
    long long idx = (long long)blockIdx.x * 256 + threadIdx.x;
    unsigned short* ptr; int rowpairs, ncols; long long pr;
    if (idx < QP)      { ptr = q; rowpairs = 1024; ncols = 2048; pr = idx; }
    else               { pr = idx - QP; if (pr >= KP) return;
                         ptr = k; rowpairs = 256;  ncols = 512; }
    int row = (int)(pr / rowpairs);
    int wi  = (int)(pr % rowpairs);
    int p   = wi & 63;
    int s   = row & 2047;
    size_t addr = (size_t)row * ncols + (size_t)(wi >> 6) * 128 + 2 * p;
    float c = fc[s * 64 + p], sn = fs[s * 64 + p];
    float xr = bf2f(ptr[addr]), xi = bf2f(ptr[addr + 1]);
    ptr[addr]     = f2bf(xr * c - xi * sn);
    ptr[addr + 1] = f2bf(xr * sn + xi * c);
}

// ---------------------------------------------------------------------------
// Flash attention, GQA (kvh = head/4), causal. Block = (b, head, 128 q rows).
// K tile staged by TDM (row-major [key][hd]); V staged transposed [hd][key]
// by vector loads so PV B-fragments stay contiguous.
// ---------------------------------------------------------------------------
__global__ __launch_bounds__(256)
void attn_kernel(const unsigned short* __restrict__ qb,
                 const unsigned short* __restrict__ kb,
                 const unsigned short* __restrict__ vb,
                 unsigned short* __restrict__ ob)
{
    __shared__ alignas(16) unsigned short Ks[32 * 128];   // [key][hd] (TDM dest)
    __shared__ alignas(16) unsigned short Vs[128 * 32];   // [hd][key]
    __shared__ alignas(16) unsigned short Ps[8][16 * 32]; // per-wave P scratch

    const int tid  = threadIdx.x;
    const int wave = tid >> 5;
    const int lane = tid & 31;
    const int hh   = lane >> 4, ml = lane & 15;
    const int q0   = blockIdx.x * 128;
    const int head = blockIdx.y;
    const int b    = blockIdx.z;
    const int kvh  = head >> 2;
    const float scale = 0.08838834764831845f;  // 1/sqrt(128)

    // Q fragments: 4 K-steps of 32 over HD=128
    Frag qa[4];
    const int qrow = q0 + wave * 16 + ml;
    const unsigned short* qrp =
        qb + ((size_t)(b * 2048 + qrow)) * 2048 + head * 128;
    #pragma unroll
    for (int ks = 0; ks < 4; ++ks) {
        qa[ks].h[0] = *(const us8*)&qrp[ks * 32 + 8 * hh];
        qa[ks].h[1] = *(const us8*)&qrp[ks * 32 + 16 + 8 * hh];
    }

    const int skey = tid >> 3;          // V staging: key 0..31
    const int shd  = (tid & 7) * 16;    // hd base

    float rmax[8], rsum[8];
    v8f oacc[8] = {};
    #pragma unroll
    for (int v = 0; v < 8; ++v) { rmax[v] = -1e30f; rsum[v] = 0.f; }

    const int nch = q0 / 32 + 4;  // causal: keys 0 .. q0+127
    for (int ch = 0; ch < nch; ++ch) {
        __syncthreads();
        // K tile via Tensor Data Mover (32 rows x 128 contiguous, stride 512)
        if (wave == 0)
            tdm_load_2d_bf16(lds_off(&Ks[0]),
                kb + ((size_t)(b * 2048 + ch * 32)) * 512 + kvh * 128,
                128, 32, 512);
        {   // V transposed by vector loads + scalar LDS scatter
            size_t src = ((size_t)(b * 2048 + ch * 32 + skey)) * 512
                         + kvh * 128 + shd;
            us8 v0v = *(const us8*)&vb[src];
            us8 v1v = *(const us8*)&vb[src + 8];
            #pragma unroll
            for (int e = 0; e < 8; ++e) {
                Vs[(shd + e) * 32 + skey]     = v0v[e];
                Vs[(shd + 8 + e) * 32 + skey] = v1v[e];
            }
        }
        if (wave == 0) __builtin_amdgcn_s_wait_tensorcnt((short)0);
        __syncthreads();

        // scores = Q @ K^T : two 16x16 subtiles over the 32 keys
        v8f s0 = {}, s1 = {};
        #pragma unroll
        for (int ks = 0; ks < 4; ++ks) {
            Frag b0, b1;
            b0.h[0] = *(const us8*)&Ks[ml * 128 + ks * 32 + 8 * hh];
            b0.h[1] = *(const us8*)&Ks[ml * 128 + ks * 32 + 16 + 8 * hh];
            b1.h[0] = *(const us8*)&Ks[(16 + ml) * 128 + ks * 32 + 8 * hh];
            b1.h[1] = *(const us8*)&Ks[(16 + ml) * 128 + ks * 32 + 16 + 8 * hh];
            s0 = __builtin_amdgcn_wmma_f32_16x16x32_bf16(false, qa[ks].v, false, b0.v, (short)0, s0, false, false);
            s1 = __builtin_amdgcn_wmma_f32_16x16x32_bf16(false, qa[ks].v, false, b1.v, (short)0, s1, false, false);
        }

        // online softmax per row (row = v + 8*hh, col = lane%16 per subtile)
        #pragma unroll
        for (int v = 0; v < 8; ++v) {
            int qpos = q0 + wave * 16 + v + 8 * hh;
            int kp0  = ch * 32 + ml;
            float a0 = s0[v] * scale + (kp0      > qpos ? -1e9f : 0.f);
            float a1 = s1[v] * scale + (kp0 + 16 > qpos ? -1e9f : 0.f);
            float mx = fmaxf(a0, a1);
            #pragma unroll
            for (int m = 1; m < 16; m <<= 1) mx = fmaxf(mx, __shfl_xor(mx, m, 32));
            float nm  = fmaxf(rmax[v], mx);
            float fac = __expf(rmax[v] - nm);
            rmax[v] = nm;
            float p0 = __expf(a0 - nm);
            float p1 = __expf(a1 - nm);
            float ps = p0 + p1;
            #pragma unroll
            for (int m = 1; m < 16; m <<= 1) ps += __shfl_xor(ps, m, 32);
            rsum[v] = rsum[v] * fac + ps;
            #pragma unroll
            for (int t = 0; t < 8; ++t) oacc[t][v] *= fac;
            Ps[wave][(v + 8 * hh) * 32 + ml]      = f2bft(p0);  // transpose via LDS
            Ps[wave][(v + 8 * hh) * 32 + 16 + ml] = f2bft(p1);
        }

        // P (16x32) @ V (32x128)
        Frag pa;
        pa.h[0] = *(const us8*)&Ps[wave][ml * 32 + 8 * hh];
        pa.h[1] = *(const us8*)&Ps[wave][ml * 32 + 16 + 8 * hh];
        #pragma unroll
        for (int t = 0; t < 8; ++t) {
            Frag bv;
            const int nrow = t * 16 + ml;
            bv.h[0] = *(const us8*)&Vs[nrow * 32 + 8 * hh];
            bv.h[1] = *(const us8*)&Vs[nrow * 32 + 16 + 8 * hh];
            oacc[t] = __builtin_amdgcn_wmma_f32_16x16x32_bf16(false, pa.v, false, bv.v, (short)0, oacc[t], false, false);
        }
    }

    // normalize + store attn (bf16) for the WO projection
    #pragma unroll
    for (int v = 0; v < 8; ++v) {
        int   row = q0 + wave * 16 + v + 8 * hh;
        float inv = 1.f / rsum[v];
        #pragma unroll
        for (int t = 0; t < 8; ++t) {
            ob[((size_t)(b * 2048 + row)) * 2048 + head * 128 + t * 16 + ml] =
                f2bf(oacc[t][v] * inv);
        }
    }
}

// ---------------------------------------------------------------------------
extern "C" void kernel_launch(void* const* d_in, const int* in_sizes, int n_in,
                              void* d_out, int out_size, void* d_ws, size_t ws_size,
                              hipStream_t stream)
{
    const float*         x  = (const float*)d_in[0];
    const float*         wq = (const float*)d_in[1];
    const float*         wk = (const float*)d_in[2];
    const float*         wv = (const float*)d_in[3];
    const float*         wo = (const float*)d_in[4];
    const float*         fc = (const float*)d_in[5];
    const float*         fs = (const float*)d_in[6];
    const unsigned char* mm = (const unsigned char*)d_in[8];  // mod_mask (bool)

    // workspace: q bf16 | k bf16 | v bf16 | attn bf16  (40 MB total)
    unsigned short* qbf = (unsigned short*)d_ws;
    unsigned short* kbf = qbf + (size_t)4096 * 2048;
    unsigned short* vbf = kbf + (size_t)4096 * 512;
    unsigned short* abf = vbf + (size_t)4096 * 512;

    // QKV projections (dual-expert, select per token)
    dual_gemm<false, true><<<dim3(64, 32), 256, 0, stream>>>(
        (const void*)x, wq, wq + (size_t)2048 * 2048, mm, (void*)qbf, 4096, 2048, 2048);
    dual_gemm<false, true><<<dim3(16, 32), 256, 0, stream>>>(
        (const void*)x, wk, wk + (size_t)2048 * 512, mm, (void*)kbf, 4096, 512, 2048);
    dual_gemm<false, true><<<dim3(16, 32), 256, 0, stream>>>(
        (const void*)x, wv, wv + (size_t)2048 * 512, mm, (void*)vbf, 4096, 512, 2048);

    // RoPE on q and k
    rope_kernel<<<20480, 256, 0, stream>>>(qbf, kbf, fc, fs);

    // flash attention (B x H x S/128 blocks)
    attn_kernel<<<dim3(16, 16, 2), 256, 0, stream>>>(qbf, kbf, vbf, abf);

    // output projection (dual-expert) -> f32 d_out
    dual_gemm<true, false><<<dim3(64, 32), 256, 0, stream>>>(
        (const void*)abf, wo, wo + (size_t)2048 * 2048, mm, d_out, 4096, 2048, 2048);
}
